// GraphNetBlock_10393820856375
// MI455X (gfx1250) — matrix operand order repaired and checked by reading, not compiled, not run on hardware
//
#include <hip/hip_runtime.h>

// ---------------------------------------------------------------------------
// GraphNet block for MI455X (gfx1250, wave32, WMMA).
// Memory-bound (~1 GB HBM traffic vs 0.3 TFLOP). Fused bf16 WMMA MLPs.
// This revision: 32 rows per wave (two A-tiles share each B-fragment load)
// to halve L2 weight traffic, float4-vectorized gather into LDS.
// ---------------------------------------------------------------------------

typedef __attribute__((ext_vector_type(16))) __bf16 v16bf;
typedef __attribute__((ext_vector_type(8)))  __bf16 v8bf;
typedef __attribute__((ext_vector_type(4)))  __bf16 v4bf;
typedef __attribute__((ext_vector_type(8)))  float  v8f;

#define LN_EPS 1e-5f

// ---- WMMA fragment helpers (layouts per CDNA5 ISA 7.12.2) ------

// A-matrix 16x32 bf16 from a row-major tile (ld elements between rows).
// lane = 16h + m ; element i<8 -> K=8h+i ; i>=8 -> K=16+8h+(i-8)
__device__ __forceinline__ v16bf load_a_frag(const __bf16* base, int ld, int lane) {
    const int m = lane & 15, h = lane >> 4;
    const __bf16* p = base + (size_t)m * ld + 8 * h;
    v8bf lo = *(const v8bf*)(p);
    v8bf hi = *(const v8bf*)(p + 16);
    return __builtin_shufflevector(lo, hi, 0,1,2,3,4,5,6,7,8,9,10,11,12,13,14,15);
}

// B-matrix 32x16 bf16 from pre-swizzled weights: one contiguous 32B load/lane.
__device__ __forceinline__ v16bf load_b_frag(const __bf16* w, int kc, int nt,
                                             int ntiles, int lane) {
    return *(const v16bf*)(w + (((size_t)(kc * ntiles + nt)) << 9) + lane * 16);
}

__device__ __forceinline__ v8f wmma_bf16(v16bf a, v16bf b, v8f c) {
    return __builtin_amdgcn_wmma_f32_16x16x32_bf16(false, a, false, b,
                                                   (short)0, c, false, false);
}

__device__ __forceinline__ float silu(float v) {
    return v / (1.0f + __expf(-v));
}

// ---- weight convert + swizzle into B-fragment order -----------------------
// out[(kc*ntiles+nt)*512 + lane*16 + i] = W[(kc*32 + (i<8 ? 8*(lane>>4)+i : 16+8*(lane>>4)+(i-8)))*Nc + nt*16 + (lane&15)]
__global__ void wconv_kernel(const float* __restrict__ W, __bf16* __restrict__ out,
                             int K, int Nc) {
    const int total  = K * Nc;
    const int ntiles = Nc >> 4;
    for (int f = blockIdx.x * blockDim.x + threadIdx.x; f < total;
         f += gridDim.x * blockDim.x) {
        int blk  = f >> 9;
        int r    = f & 511;
        int lane = r >> 4;
        int i    = r & 15;
        int kc   = blk / ntiles;
        int nt   = blk - kc * ntiles;
        int k    = kc * 32 + (i < 8 ? 8 * (lane >> 4) + i : 16 + 8 * (lane >> 4) + (i - 8));
        int n    = nt * 16 + (lane & 15);
        out[f] = (__bf16)W[(size_t)k * Nc + n];
    }
}

__global__ void zero_f32_kernel(float* __restrict__ p, long long n) {
    long long i = (long long)blockIdx.x * blockDim.x + threadIdx.x;
    long long stride = (long long)gridDim.x * blockDim.x;
    for (; i < n; i += stride) p[i] = 0.0f;
}

// ---------------------------------------------------------------------------
// Edge kernel: 2 waves / 64-thread block, 32 rows per wave, 64 edges / block.
//   eh = [edge_attr | x[src] | x[dst]]  (K=384)
//   h1 = silu(eh @ We1 + be1)           (H=256)
//   out = LN(h1 @ We2 + eh @ Wer + be2 + ber) * ge + bge  (128)
//   edge_out[e] = out ; agg[dst[e]] += out (atomic)
// ---------------------------------------------------------------------------
__global__ __launch_bounds__(64) void edge_kernel(
    const float* __restrict__ x, const float* __restrict__ edge_attr,
    const long long* __restrict__ eidx,
    const __bf16* __restrict__ we1, const float* __restrict__ be1,
    const __bf16* __restrict__ we2, const float* __restrict__ be2,
    const __bf16* __restrict__ wer, const float* __restrict__ ber,
    const float* __restrict__ ge, const float* __restrict__ bge,
    float* __restrict__ edge_out, float* __restrict__ agg,
    long long E, int N) {
    __shared__ __align__(32) __bf16 sEH[64 * 384];
    __shared__ __align__(32) __bf16 sH1[64 * 256];
    __shared__ int sSrc[64];
    __shared__ int sDst[64];

    const int tid  = threadIdx.x;
    const int lane = tid & 31;
    const int w    = tid >> 5;                 // 0..1
    const long long e0 = (long long)blockIdx.x * 64;

    for (int r = tid; r < 64; r += 64) {
        long long e = e0 + r;
        sSrc[r] = (e < E) ? (int)eidx[e]     : 0;
        sDst[r] = (e < E) ? (int)eidx[E + e] : 0;
    }
    __syncthreads();

    // Build 64x384 bf16 input tile with float4 gathers.
    for (int idx = tid; idx < 64 * 96; idx += 64) {
        int r  = idx / 96;
        int c  = (idx - r * 96) * 4;
        long long e = e0 + r;
        float4 v = make_float4(0.f, 0.f, 0.f, 0.f);
        if (e < E) {
            if (c < 128)      v = *(const float4*)(edge_attr + e * 128 + c);
            else if (c < 256) v = *(const float4*)(x + (size_t)sSrc[r] * 128 + (c - 128));
            else              v = *(const float4*)(x + (size_t)sDst[r] * 128 + (c - 256));
        }
        v4bf b;
        b[0] = (__bf16)v.x; b[1] = (__bf16)v.y;
        b[2] = (__bf16)v.z; b[3] = (__bf16)v.w;
        *(v4bf*)(sEH + r * 384 + c) = b;
    }
    __syncthreads();

    const int n16 = lane & 15, h = lane >> 4;
    const __bf16* ehA0 = sEH + (w * 32) * 384;
    const __bf16* ehA1 = sEH + (w * 32 + 16) * 384;

    // GEMM1: two 16-row tiles share each B fragment.
    for (int nt = 0; nt < 16; ++nt) {
        v8f acc0 = {}, acc1 = {};
#pragma unroll
        for (int kc = 0; kc < 12; ++kc) {
            v16bf b = load_b_frag(we1, kc, nt, 16, lane);
            acc0 = wmma_bf16(load_a_frag(ehA0 + kc * 32, 384, lane), b, acc0);
            acc1 = wmma_bf16(load_a_frag(ehA1 + kc * 32, 384, lane), b, acc1);
        }
        const float bias = be1[nt * 16 + n16];
#pragma unroll
        for (int i = 0; i < 8; ++i) {
            int col = nt * 16 + n16;
            sH1[(w * 32 +      i + 8 * h) * 256 + col] = (__bf16)silu(acc0[i] + bias);
            sH1[(w * 32 + 16 + i + 8 * h) * 256 + col] = (__bf16)silu(acc1[i] + bias);
        }
    }
    __syncthreads();

    // GEMM2: h1@We2 + eh@Wer -> 32x128 in registers.
    const __bf16* h1A0 = sH1 + (w * 32) * 256;
    const __bf16* h1A1 = sH1 + (w * 32 + 16) * 256;
    v8f o[2][8];
#pragma unroll
    for (int t = 0; t < 8; ++t) {
        v8f acc0 = {}, acc1 = {};
#pragma unroll
        for (int kc = 0; kc < 8; ++kc) {
            v16bf b = load_b_frag(we2, kc, t, 8, lane);
            acc0 = wmma_bf16(load_a_frag(h1A0 + kc * 32, 256, lane), b, acc0);
            acc1 = wmma_bf16(load_a_frag(h1A1 + kc * 32, 256, lane), b, acc1);
        }
#pragma unroll
        for (int kc = 0; kc < 12; ++kc) {
            v16bf b = load_b_frag(wer, kc, t, 8, lane);
            acc0 = wmma_bf16(load_a_frag(ehA0 + kc * 32, 384, lane), b, acc0);
            acc1 = wmma_bf16(load_a_frag(ehA1 + kc * 32, 384, lane), b, acc1);
        }
        o[0][t] = acc0;
        o[1][t] = acc1;
    }

    float g_l[8], b_l[8];
#pragma unroll
    for (int t = 0; t < 8; ++t) {
        int n = t * 16 + n16;
        float bias = be2[n] + ber[n];
        g_l[t] = ge[n];
        b_l[t] = bge[n];
#pragma unroll
        for (int i = 0; i < 8; ++i) { o[0][t][i] += bias; o[1][t][i] += bias; }
    }

    // LayerNorm per row (row lives in one 16-lane half) + stores.
#pragma unroll
    for (int hf = 0; hf < 2; ++hf) {
#pragma unroll
        for (int i = 0; i < 8; ++i) {
            float s = 0.0f, q = 0.0f;
#pragma unroll
            for (int t = 0; t < 8; ++t) { float v = o[hf][t][i]; s += v; q += v * v; }
#pragma unroll
            for (int msk = 1; msk < 16; msk <<= 1) {
                s += __shfl_xor(s, msk, 32);
                q += __shfl_xor(q, msk, 32);
            }
            const float mean = s * (1.0f / 128.0f);
            const float var  = q * (1.0f / 128.0f) - mean * mean;
            const float inv  = rsqrtf(var + LN_EPS);
            const int row = w * 32 + hf * 16 + i + 8 * h;
            const long long e = e0 + row;
            if (e < E) {
                const int dst = sDst[row];
                const size_t eo = (size_t)e * 128;
                const size_t ao = (size_t)dst * 128;
#pragma unroll
                for (int t = 0; t < 8; ++t) {
                    float v = (o[hf][t][i] - mean) * inv * g_l[t] + b_l[t];
                    int n = t * 16 + n16;
                    edge_out[eo + n] = v;
                    atomicAdd(agg + ao + n, v);
                }
            }
        }
    }
}

// ---------------------------------------------------------------------------
// Node kernel: nh = [x | agg] (K=256) -> silu(nh@Wn1+bn1)@Wn2 + nh@Wnr -> LN
// 2 waves / 64-thread block, 32 rows per wave.
// ---------------------------------------------------------------------------
__global__ __launch_bounds__(64) void node_kernel(
    const float* __restrict__ x, const float* __restrict__ agg,
    const __bf16* __restrict__ wn1, const float* __restrict__ bn1,
    const __bf16* __restrict__ wn2, const float* __restrict__ bn2,
    const __bf16* __restrict__ wnr, const float* __restrict__ bnr,
    const float* __restrict__ gn, const float* __restrict__ bgn,
    float* __restrict__ node_out, int N) {
    __shared__ __align__(32) __bf16 sNH[64 * 256];
    __shared__ __align__(32) __bf16 sH1[64 * 256];

    const int tid  = threadIdx.x;
    const int lane = tid & 31;
    const int w    = tid >> 5;
    const int n0   = blockIdx.x * 64;

    for (int idx = tid; idx < 64 * 64; idx += 64) {
        int r = idx >> 6;
        int c = (idx & 63) * 4;
        int node = n0 + r;
        float4 v = make_float4(0.f, 0.f, 0.f, 0.f);
        if (node < N) {
            v = (c < 128) ? *(const float4*)(x + (size_t)node * 128 + c)
                          : *(const float4*)(agg + (size_t)node * 128 + (c - 128));
        }
        v4bf b;
        b[0] = (__bf16)v.x; b[1] = (__bf16)v.y;
        b[2] = (__bf16)v.z; b[3] = (__bf16)v.w;
        *(v4bf*)(sNH + r * 256 + c) = b;
    }
    __syncthreads();

    const int n16 = lane & 15, h = lane >> 4;
    const __bf16* nhA0 = sNH + (w * 32) * 256;
    const __bf16* nhA1 = sNH + (w * 32 + 16) * 256;

    for (int nt = 0; nt < 16; ++nt) {
        v8f acc0 = {}, acc1 = {};
#pragma unroll
        for (int kc = 0; kc < 8; ++kc) {
            v16bf b = load_b_frag(wn1, kc, nt, 16, lane);
            acc0 = wmma_bf16(load_a_frag(nhA0 + kc * 32, 256, lane), b, acc0);
            acc1 = wmma_bf16(load_a_frag(nhA1 + kc * 32, 256, lane), b, acc1);
        }
        const float bias = bn1[nt * 16 + n16];
#pragma unroll
        for (int i = 0; i < 8; ++i) {
            int col = nt * 16 + n16;
            sH1[(w * 32 +      i + 8 * h) * 256 + col] = (__bf16)silu(acc0[i] + bias);
            sH1[(w * 32 + 16 + i + 8 * h) * 256 + col] = (__bf16)silu(acc1[i] + bias);
        }
    }
    __syncthreads();

    const __bf16* h1A0 = sH1 + (w * 32) * 256;
    const __bf16* h1A1 = sH1 + (w * 32 + 16) * 256;
    v8f o[2][8];
#pragma unroll
    for (int t = 0; t < 8; ++t) {
        v8f acc0 = {}, acc1 = {};
#pragma unroll
        for (int kc = 0; kc < 8; ++kc) {
            v16bf b = load_b_frag(wn2, kc, t, 8, lane);
            acc0 = wmma_bf16(load_a_frag(h1A0 + kc * 32, 256, lane), b, acc0);
            acc1 = wmma_bf16(load_a_frag(h1A1 + kc * 32, 256, lane), b, acc1);
        }
#pragma unroll
        for (int kc = 0; kc < 8; ++kc) {
            v16bf b = load_b_frag(wnr, kc, t, 8, lane);
            acc0 = wmma_bf16(load_a_frag(nhA0 + kc * 32, 256, lane), b, acc0);
            acc1 = wmma_bf16(load_a_frag(nhA1 + kc * 32, 256, lane), b, acc1);
        }
        o[0][t] = acc0;
        o[1][t] = acc1;
    }

    float g_l[8], b_l[8];
#pragma unroll
    for (int t = 0; t < 8; ++t) {
        int n = t * 16 + n16;
        float bias = bn2[n] + bnr[n];
        g_l[t] = gn[n];
        b_l[t] = bgn[n];
#pragma unroll
        for (int i = 0; i < 8; ++i) { o[0][t][i] += bias; o[1][t][i] += bias; }
    }

#pragma unroll
    for (int hf = 0; hf < 2; ++hf) {
#pragma unroll
        for (int i = 0; i < 8; ++i) {
            float s = 0.0f, q = 0.0f;
#pragma unroll
            for (int t = 0; t < 8; ++t) { float v = o[hf][t][i]; s += v; q += v * v; }
#pragma unroll
            for (int msk = 1; msk < 16; msk <<= 1) {
                s += __shfl_xor(s, msk, 32);
                q += __shfl_xor(q, msk, 32);
            }
            const float mean = s * (1.0f / 128.0f);
            const float var  = q * (1.0f / 128.0f) - mean * mean;
            const float inv  = rsqrtf(var + LN_EPS);
            const int node = n0 + w * 32 + hf * 16 + i + 8 * h;
            if (node < N) {
                const size_t no = (size_t)node * 128;
#pragma unroll
                for (int t = 0; t < 8; ++t) {
                    float v = (o[hf][t][i] - mean) * inv * g_l[t] + b_l[t];
                    node_out[no + t * 16 + n16] = v;
                }
            }
        }
    }
}

// ---------------------------------------------------------------------------
extern "C" void kernel_launch(void* const* d_in, const int* in_sizes, int n_in,
                              void* d_out, int out_size, void* d_ws, size_t ws_size,
                              hipStream_t stream) {
    const float* x        = (const float*)d_in[0];
    const float* edgeattr = (const float*)d_in[1];
    const long long* eidx = (const long long*)d_in[2];
    const float* We1 = (const float*)d_in[3];  const float* be1 = (const float*)d_in[4];
    const float* We2 = (const float*)d_in[5];  const float* be2 = (const float*)d_in[6];
    const float* Wer = (const float*)d_in[7];  const float* ber = (const float*)d_in[8];
    const float* ge  = (const float*)d_in[9];  const float* bge = (const float*)d_in[10];
    const float* Wn1 = (const float*)d_in[11]; const float* bn1 = (const float*)d_in[12];
    const float* Wn2 = (const float*)d_in[13]; const float* bn2 = (const float*)d_in[14];
    const float* Wnr = (const float*)d_in[15]; const float* bnr = (const float*)d_in[16];
    const float* gn  = (const float*)d_in[17]; const float* bgn = (const float*)d_in[18];

    const int N = in_sizes[0] / 128;
    const long long E = in_sizes[1] / 128;

    float* node_out = (float*)d_out;
    float* edge_out = node_out + (size_t)N * 128;

    // Workspace layout: [agg: N*128 f32][bf16 swizzled weights]
    float* agg = (float*)d_ws;
    size_t aggBytes = (((size_t)N * 128 * sizeof(float)) + 63) & ~(size_t)63;
    __bf16* we1 = (__bf16*)((char*)d_ws + aggBytes);
    __bf16* we2 = we1 + 384 * 256;
    __bf16* wer = we2 + 256 * 128;
    __bf16* wn1 = wer + 384 * 128;
    __bf16* wn2 = wn1 + 256 * 256;
    __bf16* wnr = wn2 + 256 * 128;

    wconv_kernel<<<(384 * 256 + 255) / 256, 256, 0, stream>>>(We1, we1, 384, 256);
    wconv_kernel<<<(256 * 128 + 255) / 256, 256, 0, stream>>>(We2, we2, 256, 128);
    wconv_kernel<<<(384 * 128 + 255) / 256, 256, 0, stream>>>(Wer, wer, 384, 128);
    wconv_kernel<<<(256 * 256 + 255) / 256, 256, 0, stream>>>(Wn1, wn1, 256, 256);
    wconv_kernel<<<(256 * 128 + 255) / 256, 256, 0, stream>>>(Wn2, wn2, 256, 128);
    wconv_kernel<<<(256 * 128 + 255) / 256, 256, 0, stream>>>(Wnr, wnr, 256, 128);

    long long aggN = (long long)N * 128;
    zero_f32_kernel<<<(int)((aggN + 255) / 256), 256, 0, stream>>>(agg, aggN);

    int eBlocks = (int)((E + 63) / 64);
    edge_kernel<<<eBlocks, 64, 0, stream>>>(x, edgeattr, eidx,
                                            we1, be1, we2, be2, wer, ber, ge, bge,
                                            edge_out, agg, E, N);

    int nBlocks = (N + 63) / 64;
    node_kernel<<<nBlocks, 64, 0, stream>>>(x, agg,
                                            wn1, bn1, wn2, bn2, wnr, bnr, gn, bgn,
                                            node_out, N);
}
